// MSA_19026705121969
// MI455X (gfx1250) — compile-verified, matrix operand order checked by reading, not run on hardware
//
#include <hip/hip_runtime.h>
#include <hip/hip_bf16.h>

// ---------------------------------------------------------------------------
// MSA forward for MI455X (gfx1250, wave32, WMMA).
//
// All matmuls run on v_wmma_f32_16x16x32_bf16 with bf16x3 fp32 emulation:
//   x = hi + lo (bf16 each);  acc += A_hi*B_hi + A_hi*B_lo + A_lo*B_hi (f32)
// Operands are pre-split ONCE into bf16 hi/lo planes (transposed where the
// WMMA B-layout wants K-major), so inner loops are only contiguous b128
// loads + WMMA. GEMM waves own 32x64 output tiles so each B fragment is
// amortized over two A row-fragments (load:wmma ratio 1:1).
//
// Roofline: 64.5 GFLOP total vs ~600 MB mandatory HBM traffic (403 MB of it
// the attn_weight output) + ~120 MB one-time plane prep -> ~30 us memory
// floor at 23.3 TB/s; bf16-rate matrix units keep compute comparable.
// ---------------------------------------------------------------------------

#define BDIM 8
#define SDIM 1024
#define DMODEL 768
#define HHEADS 12
#define HDIM 64

typedef __attribute__((ext_vector_type(16))) __bf16    v16bf;
typedef __attribute__((ext_vector_type(8)))  __bf16    v8bf;
typedef __attribute__((ext_vector_type(8)))  float     v8f;
typedef __attribute__((ext_vector_type(4)))  unsigned  v4u;

__device__ __forceinline__ v8f wmma_bf16(v16bf a, v16bf b, v8f c) {
  // (neg_a, A, neg_b, B, c_mod, C, reuse_a, reuse_b)
  return __builtin_amdgcn_wmma_f32_16x16x32_bf16(
      false, a, false, b, (short)0, c, false, false);
}

__device__ __forceinline__ v8f v8f_zero() {
  v8f z = {0.f, 0.f, 0.f, 0.f, 0.f, 0.f, 0.f, 0.f};
  return z;
}

__device__ __forceinline__ void split_bf16(float x, __bf16& h, __bf16& l) {
  h = (__bf16)x;
  l = (__bf16)(x - (float)h);
}

// Pack {hi,lo} bf16 pair into one uint32 (hi in [15:0], lo in [31:16]).
__device__ __forceinline__ unsigned pack_split(float x) {
  __bf16 h, l;
  split_bf16(x, h, l);
  unsigned short hu, lu;
  __builtin_memcpy(&hu, &h, 2);
  __builtin_memcpy(&lu, &l, 2);
  return (unsigned)hu | ((unsigned)lu << 16);
}

// A/B fragment from a bf16 plane: per-lane k-chunk = p[0..7] and p[16..23]
// (ISA 7.12.2 16-bit A-matrix layout). Two 16B vector loads + one shuffle.
__device__ __forceinline__ v16bf ld_frag(const __bf16* p) {
  v8bf a = *(const v8bf*)(p);
  v8bf b = *(const v8bf*)(p + 16);
  return __builtin_shufflevector(a, b, 0, 1, 2, 3, 4, 5, 6, 7,
                                 8, 9, 10, 11, 12, 13, 14, 15);
}

// Fragment from packed {hi,lo}-uint32 LDS rows: 4 x b128 loads + shuffles.
__device__ __forceinline__ void ld_frag_packed(const unsigned* p,
                                               v16bf& hi, v16bf& lo) {
  v8bf a = __builtin_bit_cast(v8bf, *(const v4u*)(p));
  v8bf b = __builtin_bit_cast(v8bf, *(const v4u*)(p + 4));
  v8bf c = __builtin_bit_cast(v8bf, *(const v4u*)(p + 16));
  v8bf d = __builtin_bit_cast(v8bf, *(const v4u*)(p + 20));
  v8bf h0 = __builtin_shufflevector(a, b, 0, 2, 4, 6, 8, 10, 12, 14);
  v8bf h1 = __builtin_shufflevector(c, d, 0, 2, 4, 6, 8, 10, 12, 14);
  v8bf l0 = __builtin_shufflevector(a, b, 1, 3, 5, 7, 9, 11, 13, 15);
  v8bf l1 = __builtin_shufflevector(c, d, 1, 3, 5, 7, 9, 11, 13, 15);
  hi = __builtin_shufflevector(h0, h1, 0, 1, 2, 3, 4, 5, 6, 7,
                               8, 9, 10, 11, 12, 13, 14, 15);
  lo = __builtin_shufflevector(l0, l1, 0, 1, 2, 3, 4, 5, 6, 7,
                               8, 9, 10, 11, 12, 13, 14, 15);
}

// ---------------------------------------------------------------------------
// Prep kernels (run once; ~120 MB total traffic, trivially memory-bound)
// ---------------------------------------------------------------------------

__global__ __launch_bounds__(256) void split_planes_kernel(
    const float* __restrict__ in, __bf16* __restrict__ h,
    __bf16* __restrict__ l, size_t n) {
  for (size_t i = (size_t)blockIdx.x * blockDim.x + threadIdx.x; i < n;
       i += (size_t)gridDim.x * blockDim.x) {
    split_bf16(in[i], h[i], l[i]);
  }
}

// W [K][N] f32 -> W^T hi/lo planes [N][K] bf16 (LDS-tiled, coalesced)
__global__ __launch_bounds__(256) void transpose_split_kernel(
    const float* __restrict__ in, __bf16* __restrict__ th,
    __bf16* __restrict__ tl, int K, int N) {
  __shared__ float tile[16][17];
  const int kb = blockIdx.y * 16, nb = blockIdx.x * 16;
  const int tx = threadIdx.x & 15, ty = threadIdx.x >> 4;
  tile[ty][tx] = in[(size_t)(kb + ty) * N + nb + tx];
  __syncthreads();
  const float x = tile[tx][ty];  // element (k = kb+tx, n = nb+ty)
  __bf16 h, l;
  split_bf16(x, h, l);
  const size_t dst = (size_t)(nb + ty) * K + kb + tx;
  th[dst] = h;
  tl[dst] = l;
}

// Extract V^T from qkv planes: [token][2D + h*64 + hd] -> [(b,h,hd)][token]
__global__ __launch_bounds__(256) void vtrans_kernel(
    const __bf16* __restrict__ qh, const __bf16* __restrict__ ql,
    __bf16* __restrict__ vth, __bf16* __restrict__ vtl) {
  __shared__ __bf16 t0[16][18];
  __shared__ __bf16 t1[16][18];
  const int bh = blockIdx.z;  // b*H + h
  const int b = bh / HHEADS, h = bh % HHEADS;
  const int tok0 = blockIdx.x * 16, hd0 = blockIdx.y * 16;
  const int tx = threadIdx.x & 15, ty = threadIdx.x >> 4;
  const size_t src = (size_t)(b * SDIM + tok0 + ty) * (3 * DMODEL) +
                     2 * DMODEL + h * HDIM + hd0 + tx;
  t0[ty][tx] = qh[src];
  t1[ty][tx] = ql[src];
  __syncthreads();
  const size_t dst = ((size_t)bh * HDIM + hd0 + ty) * SDIM + tok0 + tx;
  vth[dst] = t0[tx][ty];
  vtl[dst] = t1[tx][ty];
}

// ---------------------------------------------------------------------------
// C[M,N] = A[M,K] @ B[K,N] from pre-split bf16 planes; B given TRANSPOSED
// (Bt [N][K]) so the B-fragment is a contiguous row read. grid =
// (N/64, M/128), 4 waves/block; each wave owns a 32 x 64 tile (2 row-frags
// x 4 col-tiles = 8 accumulators) so every B fragment feeds 6 WMMAs.
// OUT_BF16: write C as hi/lo planes (feeds the next WMMA stage) else f32.
// ---------------------------------------------------------------------------
template <bool OUT_BF16>
__global__ __launch_bounds__(128) void gemm_bf16x3_kernel(
    const __bf16* __restrict__ Ah, const __bf16* __restrict__ Al,
    const __bf16* __restrict__ Bth, const __bf16* __restrict__ Btl,
    float* __restrict__ Cf, __bf16* __restrict__ Ch, __bf16* __restrict__ Cl,
    int M, int N, int K) {
  const int lane = threadIdx.x & 31;
  const int wave = threadIdx.x >> 5;
  const int rowBase = blockIdx.y * 128 + wave * 32;
  const int colBase = blockIdx.x * 64;
  const int m = lane & 15;
  const int ks = (lane >> 4) << 3;  // 0 or 8

  v8f acc[2][4];
#pragma unroll
  for (int rt = 0; rt < 2; ++rt)
#pragma unroll
    for (int nt = 0; nt < 4; ++nt) acc[rt][nt] = v8f_zero();

  const size_t aRow0 = (size_t)(rowBase + m) * K;
  const size_t aRow1 = (size_t)(rowBase + 16 + m) * K;
  for (int k0 = 0; k0 < K; k0 += 32) {
    const v16bf a0h = ld_frag(Ah + aRow0 + k0 + ks);
    const v16bf a0l = ld_frag(Al + aRow0 + k0 + ks);
    const v16bf a1h = ld_frag(Ah + aRow1 + k0 + ks);
    const v16bf a1l = ld_frag(Al + aRow1 + k0 + ks);
#pragma unroll
    for (int nt = 0; nt < 4; ++nt) {
      const size_t bRow = (size_t)(colBase + nt * 16 + m) * K + k0 + ks;
      const v16bf bhi = ld_frag(Bth + bRow);
      const v16bf blo = ld_frag(Btl + bRow);
      acc[0][nt] = wmma_bf16(a0h, bhi, acc[0][nt]);
      acc[1][nt] = wmma_bf16(a1h, bhi, acc[1][nt]);
      acc[0][nt] = wmma_bf16(a0h, blo, acc[0][nt]);
      acc[1][nt] = wmma_bf16(a1h, blo, acc[1][nt]);
      acc[0][nt] = wmma_bf16(a0l, bhi, acc[0][nt]);
      acc[1][nt] = wmma_bf16(a1l, bhi, acc[1][nt]);
    }
  }

  // C/D layout: row = v + 8*(lane>>4), col = lane&15
#pragma unroll
  for (int rt = 0; rt < 2; ++rt) {
#pragma unroll
    for (int nt = 0; nt < 4; ++nt) {
      const int n = colBase + nt * 16 + m;
#pragma unroll
      for (int v = 0; v < 8; ++v) {
        const size_t idx =
            (size_t)(rowBase + rt * 16 + v + ((lane >> 4) << 3)) * N + n;
        const float x = acc[rt][nt][v];
        if (OUT_BF16) {
          __bf16 h, l;
          split_bf16(x, h, l);
          Ch[idx] = h;
          Cl[idx] = l;
        } else {
          Cf[idx] = x;
        }
      }
    }
  }
}

// ---------------------------------------------------------------------------
// Attention: one workgroup (4 waves) per (b, h, 16-row Q tile).
//  1) scores[16,1024] (f32) -> 64 KB dynamic LDS
//  2) row softmax; probs -> attn_weight (f32, coalesced) and, IN PLACE in the
//     same 4-byte LDS slots, packed {bf16 hi, bf16 lo}
//  3) ctx = P @ V^T-planes; each wave owns one 16x16 hd tile, K-loop = 1024
// ---------------------------------------------------------------------------
__global__ __launch_bounds__(128) void attn_kernel(
    const __bf16* __restrict__ qh, const __bf16* __restrict__ ql,  // [8192][3D]
    const __bf16* __restrict__ vth, const __bf16* __restrict__ vtl,// [(b,h,hd)][S]
    float* __restrict__ attn,                                      // [B,H,S,S]
    __bf16* __restrict__ ctxh, __bf16* __restrict__ ctxl) {        // [8192][D]
  extern __shared__ unsigned char smraw[];
  float* smf = (float*)smraw;        // 16 x 1024 f32 scores / exp values
  unsigned* smu = (unsigned*)smraw;  // same slots, packed probs afterwards

  __shared__ float red[16 * 8];
  __shared__ float rowmax[16];
  __shared__ float rowinv[16];

  const int it = blockIdx.x;  // q row tile 0..63
  const int h  = blockIdx.y;  // 0..11
  const int b  = blockIdx.z;  // 0..7
  const int tid = threadIdx.x;
  const int lane = tid & 31;
  const int wave = tid >> 5;
  const int m = lane & 15;
  const int ks = (lane >> 4) << 3;

  const float scale = 0.03608439182435161f;  // 768^-0.5 (hidden-dim scaling!)
  const int threeD = 3 * DMODEL;

  // ---- Q fragments (HD=64 -> two 32-wide k-steps), loaded once per wave.
  v16bf qfh[2], qfl[2];
  {
    const size_t qRow = (size_t)(b * SDIM + it * 16 + m) * threeD + h * HDIM;
#pragma unroll
    for (int t = 0; t < 2; ++t) {
      qfh[t] = ld_frag(qh + qRow + t * 32 + ks);
      qfl[t] = ld_frag(ql + qRow + t * 32 + ks);
    }
  }

  // ---- Phase 1: scores. Each wave covers 16 of the 64 key chunks.
  for (int jc = wave; jc < 64; jc += 4) {
    v8f acc = v8f_zero();
    const size_t kRow =
        (size_t)(b * SDIM + jc * 16 + m) * threeD + DMODEL + h * HDIM;
#pragma unroll
    for (int t = 0; t < 2; ++t) {
      const v16bf khi = ld_frag(qh + kRow + t * 32 + ks);
      const v16bf klo = ld_frag(ql + kRow + t * 32 + ks);
      acc = wmma_bf16(qfh[t], khi, acc);
      acc = wmma_bf16(qfh[t], klo, acc);
      acc = wmma_bf16(qfl[t], khi, acc);
    }
#pragma unroll
    for (int v = 0; v < 8; ++v) {
      const int r = v + ((lane >> 4) << 3);
      smf[r * SDIM + jc * 16 + m] = acc[v] * scale;
    }
  }
  __syncthreads();

  // ---- Phase 2: softmax, 8 threads per row.
  {
    const int r = tid & 15;
    const int sub = tid >> 4;
    float* row = smf + r * SDIM;
    const int c0 = sub * 128;

    float lmax = -3.4e38f;
    for (int c = c0; c < c0 + 128; ++c) lmax = fmaxf(lmax, row[c]);
    red[r * 8 + sub] = lmax;
    __syncthreads();
    if (sub == 0) {
      float mx = red[r * 8];
#pragma unroll
      for (int j = 1; j < 8; ++j) mx = fmaxf(mx, red[r * 8 + j]);
      rowmax[r] = mx;
    }
    __syncthreads();
    const float mx = rowmax[r];
    float lsum = 0.f;
    for (int c = c0; c < c0 + 128; ++c) {
      const float e = __expf(row[c] - mx);
      row[c] = e;
      lsum += e;
    }
    red[r * 8 + sub] = lsum;
    __syncthreads();
    if (sub == 0) {
      float s = red[r * 8];
#pragma unroll
      for (int j = 1; j < 8; ++j) s += red[r * 8 + j];
      rowinv[r] = 1.f / s;
    }
    __syncthreads();
  }

  // Normalize: f32 probs -> global attn (coalesced) and packed bf16 pair back
  // into the SAME 4-byte LDS slot (one element per thread -> race-free).
  {
    float* attnBase = attn + ((size_t)(b * HHEADS + h) * SDIM + it * 16) * SDIM;
    for (int idx = tid; idx < 16 * SDIM; idx += 128) {
      const int r = idx >> 10;
      const int c = idx & 1023;
      const float p = smf[r * SDIM + c] * rowinv[r];
      attnBase[(size_t)r * SDIM + c] = p;
      smu[r * SDIM + c] = pack_split(p);
    }
  }
  __syncthreads();

  // ---- Phase 3: ctx = P @ V. Wave w owns hd columns [w*16, w*16+16).
  {
    v8f acc = v8f_zero();
    const int ncol = wave * 16 + m;  // 0..63 within head
    const size_t vRow = ((size_t)(b * HHEADS + h) * HDIM + ncol) * SDIM;
    for (int kk = 0; kk < SDIM; kk += 32) {
      v16bf phi, plo;
      ld_frag_packed(smu + m * SDIM + kk + ks, phi, plo);
      const v16bf vhi = ld_frag(vth + vRow + kk + ks);
      const v16bf vlo = ld_frag(vtl + vRow + kk + ks);
      acc = wmma_bf16(phi, vhi, acc);
      acc = wmma_bf16(phi, vlo, acc);
      acc = wmma_bf16(plo, vhi, acc);
    }
#pragma unroll
    for (int v = 0; v < 8; ++v) {
      const size_t idx =
          (size_t)(b * SDIM + it * 16 + v + ((lane >> 4) << 3)) * DMODEL +
          h * HDIM + ncol;
      __bf16 hh, ll;
      split_bf16(acc[v], hh, ll);
      ctxh[idx] = hh;
      ctxl[idx] = ll;
    }
  }
}

// ---------------------------------------------------------------------------
extern "C" void kernel_launch(void* const* d_in, const int* in_sizes, int n_in,
                              void* d_out, int out_size, void* d_ws,
                              size_t ws_size, hipStream_t stream) {
  (void)in_sizes; (void)n_in; (void)out_size; (void)ws_size;

  const float* x    = (const float*)d_in[0];  // [B,S,D]
  const float* Wqkv = (const float*)d_in[1];  // [D,3D]
  const float* Wout = (const float*)d_in[2];  // [D,D]

  float* out  = (float*)d_out;                       // [B,S,D]
  float* attn = out + (size_t)BDIM * SDIM * DMODEL;  // [B,H,S,S]

  const int M = BDIM * SDIM;  // 8192 token rows
  const size_t nX   = (size_t)M * DMODEL;
  const size_t nWq  = (size_t)DMODEL * 3 * DMODEL;
  const size_t nWo  = (size_t)DMODEL * DMODEL;
  const size_t nQKV = (size_t)M * 3 * DMODEL;

  // ---- workspace layout (bf16 planes)
  __bf16* p = (__bf16*)d_ws;
  __bf16* xh = p;            p += nX;
  __bf16* xl = p;            p += nX;
  __bf16* wqt_h = p;         p += nWq;   // W_qkv^T [3D][D]
  __bf16* wqt_l = p;         p += nWq;
  __bf16* wot_h = p;         p += nWo;   // W_out^T [D][D]
  __bf16* wot_l = p;         p += nWo;
  __bf16* qkv_h = p;         p += nQKV;  // [M][3D]
  __bf16* qkv_l = p;         p += nQKV;
  __bf16* vt_h = p;          p += nX;    // [(b,h,hd)][S]
  __bf16* vt_l = p;          p += nX;
  __bf16* ctx_h = p;         p += nX;    // [M][D]
  __bf16* ctx_l = p;         p += nX;

  // 0) one-time operand prep
  split_planes_kernel<<<8192, 256, 0, stream>>>(x, xh, xl, nX);
  {
    dim3 g((3 * DMODEL) / 16, DMODEL / 16);
    transpose_split_kernel<<<g, 256, 0, stream>>>(Wqkv, wqt_h, wqt_l, DMODEL,
                                                  3 * DMODEL);
  }
  {
    dim3 g(DMODEL / 16, DMODEL / 16);
    transpose_split_kernel<<<g, 256, 0, stream>>>(Wout, wot_h, wot_l, DMODEL,
                                                  DMODEL);
  }

  // 1) qkv = x @ W_qkv  (8192 x 768 x 2304), output as bf16 hi/lo planes
  {
    dim3 grid((3 * DMODEL) / 64, M / 128);
    gemm_bf16x3_kernel<true><<<grid, 128, 0, stream>>>(
        xh, xl, wqt_h, wqt_l, nullptr, qkv_h, qkv_l, M, 3 * DMODEL, DMODEL);
  }

  // 1b) V^T planes for the P@V stage
  {
    dim3 grid(SDIM / 16, HDIM / 16, BDIM * HHEADS);
    vtrans_kernel<<<grid, 256, 0, stream>>>(qkv_h, qkv_l, vt_h, vt_l);
  }

  // 2) attention: scores + softmax (-> attn output) + P@V (-> ctx planes)
  {
    dim3 grid(SDIM / 16, HHEADS, BDIM);
    const size_t shmem = (size_t)16 * SDIM * sizeof(float);  // 64 KB / 320 KB
    attn_kernel<<<grid, 128, shmem, stream>>>(qkv_h, qkv_l, vt_h, vt_l, attn,
                                              ctx_h, ctx_l);
  }

  // 3) out = ctx @ W_out  (8192 x 768 x 768), f32 output
  {
    dim3 grid(DMODEL / 64, M / 128);
    gemm_bf16x3_kernel<false><<<grid, 128, 0, stream>>>(
        ctx_h, ctx_l, wot_h, wot_l, out, nullptr, nullptr, M, DMODEL, DMODEL);
  }
}
